// _PointnetSAModuleFSBase_876173328637
// MI455X (gfx1250) — matrix operand order repaired and checked by reading, not compiled, not run on hardware
//
#include <hip/hip_runtime.h>
#include <hip/hip_bf16.h>

// ---- problem constants (from reference) ----
#define B_     2
#define N_     16384
#define S_     4096
#define NS_    32
#define CIN_   64
#define RADIUS_ 0.08f

// WMMA vector types (probe-confirmed signatures)
typedef __attribute__((ext_vector_type(16))) _Float16 v16h;
typedef __attribute__((ext_vector_type(8)))  _Float16 v8h;
typedef __attribute__((ext_vector_type(8)))  float    v8f;

// ============================================================================
// Kernel 1: Furthest Point Sampling. One 1024-thread block per batch.
// Each thread owns 16 points in registers; per-iteration block argmax via
// wave shuffles + 32-entry LDS reduction. Writes new_xyz to ws and d_out.
// ============================================================================
__global__ __launch_bounds__(1024) void fps_kernel(
    const float* __restrict__ xyz,
    float* __restrict__ newXyz,      // ws: (B,S,3)
    float* __restrict__ outXyz)      // d_out: (B,S,3)
{
    const int b   = blockIdx.x;
    const int tid = threadIdx.x;
    const float* X = xyz + (size_t)b * N_ * 3;
    const int PPT = N_ / 1024;       // 16 points per thread

    float px[16], py[16], pz[16], md[16];
#pragma unroll
    for (int k = 0; k < PPT; ++k) {
        int p = tid + k * 1024;
        px[k] = X[p * 3 + 0];
        py[k] = X[p * 3 + 1];
        pz[k] = X[p * 3 + 2];
        md[k] = 1e10f;
    }

    __shared__ float cur[3];
    __shared__ float sVal[32];
    __shared__ int   sIdx[32];

    if (tid == 0) {
        // first sample = index 0 (reference semantics)
        float x0 = X[0], y0 = X[1], z0 = X[2];
        newXyz[(size_t)(b * S_ + 0) * 3 + 0] = x0;
        newXyz[(size_t)(b * S_ + 0) * 3 + 1] = y0;
        newXyz[(size_t)(b * S_ + 0) * 3 + 2] = z0;
        outXyz[(size_t)(b * S_ + 0) * 3 + 0] = x0;
        outXyz[(size_t)(b * S_ + 0) * 3 + 1] = y0;
        outXyz[(size_t)(b * S_ + 0) * 3 + 2] = z0;
        cur[0] = x0; cur[1] = y0; cur[2] = z0;
    }
    __syncthreads();

    for (int it = 1; it < S_; ++it) {
        float cx = cur[0], cy = cur[1], cz = cur[2];
        float bestV = -1.0f;
        int   bestI = 0;
#pragma unroll
        for (int k = 0; k < PPT; ++k) {
            float dx = px[k] - cx, dy = py[k] - cy, dz = pz[k] - cz;
            float d  = dx * dx + dy * dy + dz * dz;
            float nm = fminf(md[k], d);
            md[k] = nm;
            if (nm > bestV) { bestV = nm; bestI = tid + k * 1024; }
        }
        // wave32 reduction (first-max: prefer lower index on ties)
#pragma unroll
        for (int off = 16; off > 0; off >>= 1) {
            float ov = __shfl_xor(bestV, off);
            int   oi = __shfl_xor(bestI, off);
            if (ov > bestV || (ov == bestV && oi < bestI)) { bestV = ov; bestI = oi; }
        }
        int wv = tid >> 5;
        if ((tid & 31) == 0) { sVal[wv] = bestV; sIdx[wv] = bestI; }
        __syncthreads();
        if (tid < 32) {
            bestV = sVal[tid]; bestI = sIdx[tid];
#pragma unroll
            for (int off = 16; off > 0; off >>= 1) {
                float ov = __shfl_xor(bestV, off);
                int   oi = __shfl_xor(bestI, off);
                if (ov > bestV || (ov == bestV && oi < bestI)) { bestV = ov; bestI = oi; }
            }
            if (tid == 0) sIdx[0] = bestI;
        }
        __syncthreads();
        int sel = sIdx[0];
        if (tid == (sel & 1023)) {  // owning thread publishes coords + outputs
            int k = sel >> 10;
            cur[0] = px[k]; cur[1] = py[k]; cur[2] = pz[k];
            newXyz[(size_t)(b * S_ + it) * 3 + 0] = px[k];
            newXyz[(size_t)(b * S_ + it) * 3 + 1] = py[k];
            newXyz[(size_t)(b * S_ + it) * 3 + 2] = pz[k];
            outXyz[(size_t)(b * S_ + it) * 3 + 0] = px[k];
            outXyz[(size_t)(b * S_ + it) * 3 + 1] = py[k];
            outXyz[(size_t)(b * S_ + it) * 3 + 2] = pz[k];
        }
        __syncthreads();
    }
}

// ============================================================================
// Kernel 2: Ball query. One wave32 per query point; scan N points in chunks
// of 32, ballot + prefix popcount appends the first NS in-radius indices in
// ascending order; pad with first found index (reference semantics).
// ============================================================================
__global__ __launch_bounds__(256) void ball_kernel(
    const float* __restrict__ xyz,
    const float* __restrict__ newXyz,
    int* __restrict__ ballIdx,       // (B,S,NS)
    int* __restrict__ ballCnt)       // (B,S)
{
    const int gw   = (blockIdx.x * 256 + threadIdx.x) >> 5;  // global wave id
    const int lane = threadIdx.x & 31;
    const int b = gw / S_;
    const int s = gw % S_;

    const float cx = newXyz[(size_t)(b * S_ + s) * 3 + 0];
    const float cy = newXyz[(size_t)(b * S_ + s) * 3 + 1];
    const float cz = newXyz[(size_t)(b * S_ + s) * 3 + 2];
    const float* X = xyz + (size_t)b * N_ * 3;
    const float r2 = RADIUS_ * RADIUS_;

    int cnt = 0, first = -1;
    int* dst = ballIdx + (size_t)(b * S_ + s) * NS_;

    for (int base = 0; base < N_ && cnt < NS_; base += 32) {
        int p = base + lane;
        float dx = X[p * 3 + 0] - cx;
        float dy = X[p * 3 + 1] - cy;
        float dz = X[p * 3 + 2] - cz;
        bool valid = (dx * dx + dy * dy + dz * dz) < r2;
        unsigned mask = (unsigned)__ballot(valid);
        if (first < 0 && mask) first = base + __builtin_ctz(mask);
        if (valid) {
            int slot = cnt + __popc(mask & ((1u << lane) - 1u));
            if (slot < NS_) dst[slot] = p;
        }
        cnt += __popc(mask);
    }
    int c = cnt < NS_ ? cnt : NS_;
    if (first < 0) first = 0;
    if (lane >= c) dst[lane] = first;          // pad with first index
    if (lane == 0) ballCnt[b * S_ + s] = cnt;
}

// ============================================================================
// Kernel 3: fused group + 3-layer shared MLP (f16 WMMA, f32 accum) + masked
// max-pool. Tile = 128 pixels = 4 balls x 32 neighbors. 8 waves; wave w owns
// N-tile of 16 pixels. Weights f16 (zero-padded K) in LDS; activations
// pixel-major in LDS so WMMA B operands are contiguous 32B chunks.
// ============================================================================
#define TPIX  128
#define K0PAD 96   // layer0 K: 67 -> 96 (3 k-steps of 32)

// A operand: 16x32 f16 tile of weights W[row][k], row-major with stride kpad.
// Contiguous k order shared with load_B: lane group h holds K = h*16 + i  (16 contiguous).
__device__ inline v16h load_A(const _Float16* w, int row, int kbase, int kpad, int half) {
    const v8h lo = *(const v8h*)&w[row * kpad + kbase + half * 16];
    const v8h hi = *(const v8h*)&w[row * kpad + kbase + half * 16 + 8];
    v16h a;
#pragma unroll
    for (int i = 0; i < 8; ++i) { a[i] = lo[i]; a[8 + i] = hi[i]; }
    return a;
}

// B operand: 32x16 f16 tile of activations act[pixel][k], pixel-major.
// Contiguous k order shared with load_A: lane group h holds K = h*16 + i  (16 contiguous).
__device__ inline v16h load_B(const _Float16* act, int pix, int kbase, int kpad, int half) {
    return *(const v16h*)&act[pix * kpad + kbase + half * 16];
}

__global__ __launch_bounds__(256) void fused_mlp_kernel(
    const float* __restrict__ xyz,  const float* __restrict__ feat,
    const float* __restrict__ W0,   const float* __restrict__ b0,
    const float* __restrict__ W1,   const float* __restrict__ b1,
    const float* __restrict__ W2,   const float* __restrict__ b2,
    const float* __restrict__ newXyz,
    const int*   __restrict__ ballIdx,
    const int*   __restrict__ ballCnt,
    float* __restrict__ outFeat)     // d_out + B*S*3 : (B,128,S)
{
    __shared__ __attribute__((aligned(32))) _Float16 sW0[64 * K0PAD];   // 12 KB
    __shared__ __attribute__((aligned(32))) _Float16 sW1[64 * 64];      //  8 KB
    __shared__ __attribute__((aligned(32))) _Float16 sW2[128 * 64];     // 16 KB
    __shared__ __attribute__((aligned(32))) _Float16 sActA[TPIX * K0PAD]; // 24 KB (reused as layer2 input)
    __shared__ __attribute__((aligned(32))) _Float16 sAct1[TPIX * 64];    // 16 KB
    __shared__ __attribute__((aligned(32))) float    sFin[TPIX * 128];    // 64 KB

    const int tid   = threadIdx.x;
    const int tile  = blockIdx.x;              // B * S/4 tiles
    const int b     = tile / (S_ / 4);
    const int qbase = (tile % (S_ / 4)) * 4;

    // ---- stage 1: weights -> LDS f16, zero-padded K for layer 0 ----
    for (int t = tid; t < 64 * K0PAD; t += 256) {
        int o = t / K0PAD, c = t % K0PAD;
        sW0[t] = (_Float16)((c < 67) ? W0[o * 67 + c] : 0.0f);
    }
    for (int t = tid; t < 64 * 64;  t += 256) sW1[t] = (_Float16)W1[t];
    for (int t = tid; t < 128 * 64; t += 256) sW2[t] = (_Float16)W2[t];

    // ---- stage 2: gather input (rel xyz + features) -> sActA[pix][96] ----
    {
        int p = tid >> 1, part = tid & 1;      // 2 threads per pixel
        int ball = p >> 5, j = p & 31;
        int q = qbase + ball;
        int n = ballIdx[(size_t)(b * S_ + q) * NS_ + j];
        _Float16* dst = &sActA[p * K0PAD];
        if (part == 0) {
            const float* cx = &newXyz[(size_t)(b * S_ + q) * 3];
            dst[0] = (_Float16)(xyz[(size_t)(b * N_ + n) * 3 + 0] - cx[0]);
            dst[1] = (_Float16)(xyz[(size_t)(b * N_ + n) * 3 + 1] - cx[1]);
            dst[2] = (_Float16)(xyz[(size_t)(b * N_ + n) * 3 + 2] - cx[2]);
#pragma unroll 5
            for (int c = 0; c < 45; ++c)
                dst[3 + c] = (_Float16)feat[(size_t)(b * CIN_ + c) * N_ + n];
        } else {
#pragma unroll 5
            for (int c = 45; c < CIN_; ++c)
                dst[3 + c] = (_Float16)feat[(size_t)(b * CIN_ + c) * N_ + n];
            for (int c = 3 + CIN_; c < K0PAD; ++c) dst[c] = (_Float16)0.0f;
        }
    }
    __syncthreads();

    const int lane = tid & 31;
    const int wv   = tid >> 5;
    const int half = lane >> 4;
    const int r    = lane & 15;
    const int pix  = wv * 16 + r;              // this wave's N-tile column

    // ---- layer 0: (67->64), K=96, 3 k-steps ----
    for (int m = 0; m < 4; ++m) {
        v8f acc = {};
#pragma unroll
        for (int ks = 0; ks < 3; ++ks) {
            v16h a  = load_A(sW0, m * 16 + r, ks * 32, K0PAD, half);
            v16h bm = load_B(sActA, pix, ks * 32, K0PAD, half);
            acc = __builtin_amdgcn_wmma_f32_16x16x32_f16(
                      false, a, false, bm, (short)0, acc, false, false);
        }
        int chb = m * 16 + half * 8;
        v8h o;
#pragma unroll
        for (int j = 0; j < 8; ++j) {
            float v = acc[j] + b0[chb + j];
            o[j] = (_Float16)(v > 0.0f ? v : 0.0f);
        }
        *(v8h*)&sAct1[pix * 64 + chb] = o;
    }
    __syncthreads();

    // ---- layer 1: (64->64), K=64, 2 k-steps; output reuses sActA storage ----
    _Float16* sAct2 = sActA;
    for (int m = 0; m < 4; ++m) {
        v8f acc = {};
#pragma unroll
        for (int ks = 0; ks < 2; ++ks) {
            v16h a  = load_A(sW1, m * 16 + r, ks * 32, 64, half);
            v16h bm = load_B(sAct1, pix, ks * 32, 64, half);
            acc = __builtin_amdgcn_wmma_f32_16x16x32_f16(
                      false, a, false, bm, (short)0, acc, false, false);
        }
        int chb = m * 16 + half * 8;
        v8h o;
#pragma unroll
        for (int j = 0; j < 8; ++j) {
            float v = acc[j] + b1[chb + j];
            o[j] = (_Float16)(v > 0.0f ? v : 0.0f);
        }
        *(v8h*)&sAct2[pix * 64 + chb] = o;
    }
    __syncthreads();

    // ---- layer 2: (64->128), K=64, 2 k-steps; f32 out -> sFin[pix][128] ----
    for (int m = 0; m < 8; ++m) {
        v8f acc = {};
#pragma unroll
        for (int ks = 0; ks < 2; ++ks) {
            v16h a  = load_A(sW2, m * 16 + r, ks * 32, 64, half);
            v16h bm = load_B(sAct2, pix, ks * 32, 64, half);
            acc = __builtin_amdgcn_wmma_f32_16x16x32_f16(
                      false, a, false, bm, (short)0, acc, false, false);
        }
        int chb = m * 16 + half * 8;
#pragma unroll
        for (int j = 0; j < 8; ++j) {
            float v = acc[j] + b2[chb + j];
            sFin[pix * 128 + chb + j] = (v > 0.0f ? v : 0.0f);
        }
    }
    __syncthreads();

    // ---- masked max-pool over NS=32 neighbors: 4 balls x 128 channels ----
    for (int t = tid; t < 4 * 128; t += 256) {
        int ball = t >> 7, ch = t & 127;
        int q = qbase + ball;
        const float* src = &sFin[ball * 32 * 128 + ch];
        float m = 0.0f;                        // post-ReLU values are >= 0
#pragma unroll
        for (int j = 0; j < 32; ++j) m = fmaxf(m, src[j * 128]);
        if (ballCnt[b * S_ + q] == 0) m = 0.0f;
        outFeat[(size_t)(b * 128 + ch) * S_ + q] = m;
    }
}

// ============================================================================
extern "C" void kernel_launch(void* const* d_in, const int* in_sizes, int n_in,
                              void* d_out, int out_size, void* d_ws, size_t ws_size,
                              hipStream_t stream) {
    const float* xyz  = (const float*)d_in[0];
    const float* feat = (const float*)d_in[1];
    const float* W0   = (const float*)d_in[2];
    const float* b0   = (const float*)d_in[3];
    const float* W1   = (const float*)d_in[4];
    const float* b1   = (const float*)d_in[5];
    const float* W2   = (const float*)d_in[6];
    const float* b2   = (const float*)d_in[7];

    float* out     = (float*)d_out;
    float* outXyz  = out;                        // (B,S,3)
    float* outFeat = out + (size_t)B_ * S_ * 3;  // (B,128,S)

    char*  w       = (char*)d_ws;
    float* newXyz  = (float*)w;  w += (size_t)B_ * S_ * 3 * sizeof(float);
    int*   ballIdx = (int*)w;    w += (size_t)B_ * S_ * NS_ * sizeof(int);
    int*   ballCnt = (int*)w;    w += (size_t)B_ * S_ * sizeof(int);

    fps_kernel<<<B_, 1024, 0, stream>>>(xyz, newXyz, outXyz);

    ball_kernel<<<(B_ * S_ * 32) / 256, 256, 0, stream>>>(xyz, newXyz, ballIdx, ballCnt);

    fused_mlp_kernel<<<B_ * (S_ / 4), 256, 0, stream>>>(
        xyz, feat, W0, b0, W1, b1, W2, b2, newXyz, ballIdx, ballCnt, outFeat);
}
